// GroupedQueryAttention_55645596287144
// MI455X (gfx1250) — compile-verified, matrix operand order checked by reading, not run on hardware
//
#include <hip/hip_runtime.h>
#include <hip/hip_bf16.h>
#include <stdint.h>

#define T_SEQ  2048
#define C_DIM  2048
#define NH     32
#define NKV    8
#define DH     64
#define WIN    512
#define QKVN   3072   // 2048 q + 512 k + 512 v

typedef __attribute__((ext_vector_type(16))) __bf16 v16bf;
typedef __attribute__((ext_vector_type(8)))  __bf16 v8bf;
typedef __attribute__((ext_vector_type(8)))  float  v8f;

static __device__ __forceinline__ __bf16 f2bf(float f) {
    union { float f; uint32_t u; } v; v.f = f;
    uint32_t r = (v.u + 0x7fffu + ((v.u >> 16) & 1u)) >> 16;
    union { uint16_t s; __bf16 b; } o; o.s = (uint16_t)r;
    return o.b;
}

// Build a v16bf WMMA fragment (ISA 7.12.2): lane holds K = base..base+7 and
// 16+base..16+base+7, base = (lane>>4)*8.
static __device__ __forceinline__ v16bf load_frag(const __bf16* row, int k0, int base) {
    v8bf lo = *(const v8bf*)(row + k0 + base);
    v8bf hi = *(const v8bf*)(row + k0 + 16 + base);
    v16bf r;
#pragma unroll
    for (int i = 0; i < 8; i++) { r[i] = lo[i]; r[i + 8] = hi[i]; }
    return r;
}

// ---- CDNA5 async global->LDS (ASYNCcnt) ----
static __device__ __forceinline__ void async_b128(uint32_t lds_off, const void* gaddr) {
    asm volatile("global_load_async_to_lds_b128 %0, %1, off"
                 :: "v"(lds_off), "v"(gaddr) : "memory");
}
template <int N>
static __device__ __forceinline__ void wait_asynccnt() {
    asm volatile("s_wait_asynccnt %0" :: "i"(N) : "memory");
}

// ---------------- elementwise / layout kernels ----------------

__global__ void cast_f32_bf16(const float* __restrict__ in, __bf16* __restrict__ out, int n) {
    int i = blockIdx.x * blockDim.x + threadIdx.x;
    if (i < n) out[i] = f2bf(in[i]);
}

// out[n][k] = bf16(in[k][n]) ; in is K x N row-major (K,N multiples of 32)
__global__ void transpose_cast(const float* __restrict__ in, __bf16* __restrict__ out,
                               int K, int N) {
    __shared__ float tile[32][33];
    int n0 = blockIdx.x * 32, k0 = blockIdx.y * 32;
    int tx = threadIdx.x, ty = threadIdx.y;
#pragma unroll
    for (int i = 0; i < 4; i++) {
        int k = k0 + ty + i * 8;
        tile[ty + i * 8][tx] = in[(size_t)k * N + n0 + tx];
    }
    __syncthreads();
#pragma unroll
    for (int i = 0; i < 4; i++) {
        int n = n0 + ty + i * 8;
        out[(size_t)n * K + k0 + tx] = f2bf(tile[tx][ty + i * 8]);
    }
}

// RoPE on q: qkv[t][h*64 + j] (fp32, cols 0..2047) -> Qb[h][t][j] bf16
__global__ void rope_q_kernel(const float* __restrict__ qkv, __bf16* __restrict__ Qb) {
    int idx = blockIdx.x * blockDim.x + threadIdx.x;
    if (idx >= T_SEQ * NH * 32) return;
    int j = idx & 31;
    int h = (idx >> 5) & (NH - 1);
    int t = idx >> 10;
    float inv = powf(10000.f, -(float)(2 * j) / 64.f);
    float ang = (float)t * inv;
    float c = cosf(ang), s = sinf(ang);
    const float* src = qkv + (size_t)t * QKVN + h * 64;
    float x1 = src[j], x2 = src[j + 32];
    __bf16* dst = Qb + ((size_t)h * T_SEQ + t) * DH;
    dst[j]      = f2bf(x1 * c - x2 * s);
    dst[j + 32] = f2bf(x2 * c + x1 * s);
}

// RoPE on k: qkv cols 2048..2559 -> Kb[kv][t][j] bf16
__global__ void rope_k_kernel(const float* __restrict__ qkv, __bf16* __restrict__ Kb) {
    int idx = blockIdx.x * blockDim.x + threadIdx.x;
    if (idx >= T_SEQ * NKV * 32) return;
    int j = idx & 31;
    int h = (idx >> 5) & (NKV - 1);
    int t = idx >> 8;
    float inv = powf(10000.f, -(float)(2 * j) / 64.f);
    float ang = (float)t * inv;
    float c = cosf(ang), s = sinf(ang);
    const float* src = qkv + (size_t)t * QKVN + 2048 + h * 64;
    float x1 = src[j], x2 = src[j + 32];
    __bf16* dst = Kb + ((size_t)h * T_SEQ + t) * DH;
    dst[j]      = f2bf(x1 * c - x2 * s);
    dst[j + 32] = f2bf(x2 * c + x1 * s);
}

// V transpose: qkv cols 2560..3071 -> Vt[kv][d][t] bf16
__global__ void pack_v_kernel(const float* __restrict__ qkv, __bf16* __restrict__ Vt) {
    int idx = blockIdx.x * blockDim.x + threadIdx.x;
    if (idx >= T_SEQ * NKV * DH) return;
    int d = idx & 63;
    int h = (idx >> 6) & (NKV - 1);
    int t = idx >> 9;
    Vt[((size_t)h * DH + d) * T_SEQ + t] = f2bf(qkv[(size_t)t * QKVN + 2560 + h * 64 + d]);
}

// ---------------- LDS-staged GEMM with async double-buffering ----------------
// C(MxN,f32) = A(MxK,bf16 row-major) * Bt(NxK,bf16).  Block = 256 threads
// (8 waves) computing a 128x128 tile; per k-step (k32) A/B tiles (8 KB each)
// are staged to LDS with GLOBAL_LOAD_ASYNC_TO_LDS_B128, double-buffered with
// ONE barrier per k-step (issue happens after the data-ready barrier, which
// also proves all waves finished reading the target buffer).

static __device__ __forceinline__ void stage_tile(const __bf16* gbase, int ldk, int k0,
                                                  __bf16 (*lbuf)[32], int tid) {
#pragma unroll
    for (int j = 0; j < 2; j++) {
        int id = tid * 2 + j;
        int r = id >> 2, part = id & 3;
        const void* g = gbase + (size_t)r * ldk + k0 + part * 8;
        uint32_t l = (uint32_t)(uintptr_t)(&lbuf[r][part * 8]);
        async_b128(l, g);
    }
}

__global__ void __launch_bounds__(256) gemm_bf16_lds(
    const __bf16* __restrict__ A, const __bf16* __restrict__ Bt,
    float* __restrict__ Cmat, int K, int ldc)
{
    __shared__ __align__(16) __bf16 Abuf[2][128][32];
    __shared__ __align__(16) __bf16 Bbuf[2][128][32];

    int tid  = threadIdx.x;
    int lane = tid & 31;
    int wave = tid >> 5;
    int col  = lane & 15;
    int half = lane >> 4;
    int base = half * 8;
    int mw = wave & 3;      // 4 m-subtiles of 32 rows
    int nw = wave >> 2;     // 2 n-subtiles of 64 cols
    int m0 = blockIdx.y * 128;
    int n0 = blockIdx.x * 128;

    const __bf16* Ablk = A  + (size_t)m0 * K;
    const __bf16* Bblk = Bt + (size_t)n0 * K;

    v8f acc[2][4] = {};

    auto compute_step = [&](int buf) {
#pragma unroll
        for (int i = 0; i < 2; i++) {
            v16bf af = load_frag(&Abuf[buf][mw * 32 + i * 16 + col][0], 0, base);
#pragma unroll
            for (int j = 0; j < 4; j++) {
                v16bf bf_ = load_frag(&Bbuf[buf][nw * 64 + j * 16 + col][0], 0, base);
                acc[i][j] = __builtin_amdgcn_wmma_f32_16x16x32_bf16(
                    false, af, false, bf_, (short)0, acc[i][j], false, false);
            }
        }
    };

    // prologue: stage k-step 0 into buffer 0
    stage_tile(Ablk, K, 0, Abuf[0], tid);
    stage_tile(Bblk, K, 0, Bbuf[0], tid);

    int nsteps = K >> 5;
    int it = 0;
    for (; it < nsteps - 1; ++it) {          // branch-free hot loop
        int buf = it & 1;
        wait_asynccnt<0>();                  // own 4 ops of current tile done
        __syncthreads();                     // whole tile visible; prev reads done
        stage_tile(Ablk, K, (it + 1) << 5, Abuf[buf ^ 1], tid);
        stage_tile(Bblk, K, (it + 1) << 5, Bbuf[buf ^ 1], tid);
        compute_step(buf);                   // overlaps next-tile async copies
    }
    wait_asynccnt<0>();
    __syncthreads();
    compute_step(it & 1);                    // peeled last k-step

#pragma unroll
    for (int i = 0; i < 2; i++) {
#pragma unroll
        for (int r = 0; r < 8; r++) {
            int row = m0 + mw * 32 + i * 16 + r + 8 * half;
            float* crow = Cmat + (size_t)row * ldc + n0 + nw * 64;
#pragma unroll
            for (int j = 0; j < 4; j++) crow[j * 16 + col] = acc[i][j][r];
        }
    }
}

// ---------------- sliding-window flash attention ----------------
// Denominator trick: a 5th accumulator fed by an all-ones B fragment computes
// row sums of P via WMMA (same alpha-rescaling chain), eliminating the
// cross-lane sum reduction; every column holds the identical row sum.
__global__ void __launch_bounds__(128) attn_wmma(
    const __bf16* __restrict__ Q, const __bf16* __restrict__ Kc,
    const __bf16* __restrict__ Vt, __bf16* __restrict__ Out)
{
    __shared__ __align__(16) __bf16 Pb[4][16][32];
    int lane = threadIdx.x & 31;
    int wave = threadIdx.x >> 5;
    int col  = lane & 15;
    int half = lane >> 4;
    int base = half * 8;
    int qt = blockIdx.x * 4 + wave;
    int h  = blockIdx.y;
    int hk = h >> 2;               // n_rep = 4
    int q0 = qt * 16;
    const float scale = 0.125f;    // 1/sqrt(64)

    const __bf16* qrow = Q + ((size_t)h * T_SEQ + q0 + col) * DH;
    v16bf aq0 = load_frag(qrow, 0,  base);
    v16bf aq1 = load_frag(qrow, 32, base);

    v16bf pone;
#pragma unroll
    for (int i = 0; i < 16; i++) pone[i] = f2bf(1.0f);

    v8f o0{}, o1{}, o2{}, o3{}, osum{};
    float mrow[8];
#pragma unroll
    for (int r = 0; r < 8; r++) mrow[r] = -1e30f;

    int kb_begin = (q0 > WIN) ? ((q0 - WIN) & ~31) : 0;
    for (int kb = kb_begin; kb < q0 + 16; kb += 32) {
        const __bf16* krow0 = Kc + ((size_t)hk * T_SEQ + kb + col) * DH;
        const __bf16* krow1 = krow0 + (size_t)16 * DH;
        v16bf bk0a = load_frag(krow0, 0,  base);
        v16bf bk0b = load_frag(krow0, 32, base);
        v16bf bk1a = load_frag(krow1, 0,  base);
        v16bf bk1b = load_frag(krow1, 32, base);
        v8f z{};
        v8f s0 = __builtin_amdgcn_wmma_f32_16x16x32_bf16(false, aq0, false, bk0a, (short)0, z,  false, false);
        s0     = __builtin_amdgcn_wmma_f32_16x16x32_bf16(false, aq1, false, bk0b, (short)0, s0, false, false);
        v8f s1 = __builtin_amdgcn_wmma_f32_16x16x32_bf16(false, aq0, false, bk1a, (short)0, z,  false, false);
        s1     = __builtin_amdgcn_wmma_f32_16x16x32_bf16(false, aq1, false, bk1b, (short)0, s1, false, false);

#pragma unroll
        for (int r = 0; r < 8; r++) {
            int qi  = q0 + r + 8 * half;
            int ki0 = kb + col;
            int ki1 = ki0 + 16;
            float v0 = (ki0 <= qi && (qi - ki0) <= WIN) ? s0[r] * scale : -1e30f;
            float v1 = (ki1 <= qi && (qi - ki1) <= WIN) ? s1[r] * scale : -1e30f;
            float rm = fmaxf(v0, v1);
#pragma unroll
            for (int m = 1; m < 16; m <<= 1) rm = fmaxf(rm, __shfl_xor(rm, m, 32));
            float mnew  = fmaxf(mrow[r], rm);
            float alpha = __expf(mrow[r] - mnew);
            mrow[r] = mnew;
            float e0 = __expf(v0 - mnew);
            float e1 = __expf(v1 - mnew);
            o0[r] *= alpha; o1[r] *= alpha; o2[r] *= alpha; o3[r] *= alpha;
            osum[r] *= alpha;
            Pb[wave][r + 8 * half][col]      = f2bf(e0);
            Pb[wave][r + 8 * half][col + 16] = f2bf(e1);
        }

        // O += P * V ; row-sum += P * ones  (P via LDS D-layout -> A-layout)
        v16bf ap = load_frag(&Pb[wave][col][0], 0, base);
        const __bf16* vr0 = Vt + ((size_t)hk * DH +  0 + col) * T_SEQ;
        const __bf16* vr1 = Vt + ((size_t)hk * DH + 16 + col) * T_SEQ;
        const __bf16* vr2 = Vt + ((size_t)hk * DH + 32 + col) * T_SEQ;
        const __bf16* vr3 = Vt + ((size_t)hk * DH + 48 + col) * T_SEQ;
        v16bf bv0 = load_frag(vr0, kb, base);
        v16bf bv1 = load_frag(vr1, kb, base);
        v16bf bv2 = load_frag(vr2, kb, base);
        v16bf bv3 = load_frag(vr3, kb, base);
        o0   = __builtin_amdgcn_wmma_f32_16x16x32_bf16(false, ap, false, bv0,  (short)0, o0,   false, false);
        o1   = __builtin_amdgcn_wmma_f32_16x16x32_bf16(false, ap, false, bv1,  (short)0, o1,   false, false);
        o2   = __builtin_amdgcn_wmma_f32_16x16x32_bf16(false, ap, false, bv2,  (short)0, o2,   false, false);
        o3   = __builtin_amdgcn_wmma_f32_16x16x32_bf16(false, ap, false, bv3,  (short)0, o3,   false, false);
        osum = __builtin_amdgcn_wmma_f32_16x16x32_bf16(false, ap, false, pone, (short)0, osum, false, false);
    }

#pragma unroll
    for (int r = 0; r < 8; r++) {
        float inv = 1.0f / osum[r];
        int t = q0 + r + 8 * half;
        __bf16* orow = Out + (size_t)t * C_DIM + h * DH;
        orow[ 0 + col] = f2bf(o0[r] * inv);
        orow[16 + col] = f2bf(o1[r] * inv);
        orow[32 + col] = f2bf(o2[r] * inv);
        orow[48 + col] = f2bf(o3[r] * inv);
    }
}

// ---------------- launch ----------------

extern "C" void kernel_launch(void* const* d_in, const int* in_sizes, int n_in,
                              void* d_out, int out_size, void* d_ws, size_t ws_size,
                              hipStream_t stream)
{
    (void)in_sizes; (void)n_in; (void)out_size; (void)ws_size;
    const float* x  = (const float*)d_in[0];
    const float* wq = (const float*)d_in[1];
    const float* wk = (const float*)d_in[2];
    const float* wv = (const float*)d_in[3];
    const float* wo = (const float*)d_in[4];
    float* out = (float*)d_out;

    char* ws = (char*)d_ws;
    size_t off = 0;
    __bf16* xb   = (__bf16*)(ws + off); off += (size_t)T_SEQ * C_DIM * 2;    // 8 MiB
    __bf16* wT   = (__bf16*)(ws + off); off += (size_t)QKVN  * C_DIM * 2;    // 12 MiB
    __bf16* woT  = (__bf16*)(ws + off); off += (size_t)C_DIM * C_DIM * 2;    // 8 MiB
    float*  qkv  = (float*) (ws + off); off += (size_t)T_SEQ * QKVN  * 4;    // 24 MiB
    __bf16* Qb   = (__bf16*)(ws + off); off += (size_t)NH  * T_SEQ * DH * 2; // 8 MiB
    __bf16* Kb   = (__bf16*)(ws + off); off += (size_t)NKV * T_SEQ * DH * 2; // 2 MiB
    __bf16* Vtb  = (__bf16*)(ws + off); off += (size_t)NKV * DH * T_SEQ * 2; // 2 MiB
    __bf16* attn = (__bf16*)(ws + off); off += (size_t)T_SEQ * C_DIM * 2;    // 8 MiB

    {   int n = T_SEQ * C_DIM;
        cast_f32_bf16<<<(n + 255) / 256, 256, 0, stream>>>(x, xb, n); }

    {   dim3 blk(32, 8);
        transpose_cast<<<dim3(C_DIM / 32, C_DIM / 32), blk, 0, stream>>>(wq, wT, C_DIM, C_DIM);
        transpose_cast<<<dim3(512 / 32,  C_DIM / 32), blk, 0, stream>>>(wk, wT + (size_t)2048 * C_DIM, C_DIM, 512);
        transpose_cast<<<dim3(512 / 32,  C_DIM / 32), blk, 0, stream>>>(wv, wT + (size_t)2560 * C_DIM, C_DIM, 512);
        transpose_cast<<<dim3(C_DIM / 32, C_DIM / 32), blk, 0, stream>>>(wo, woT, C_DIM, C_DIM); }

    // fused QKV projection: [2048 x 2048] x [2048 x 3072] -> fp32
    gemm_bf16_lds<<<dim3(QKVN / 128, T_SEQ / 128), 256, 0, stream>>>(
        xb, wT, qkv, C_DIM, QKVN);

    {   int n = T_SEQ * NH * 32;
        rope_q_kernel<<<(n + 255) / 256, 256, 0, stream>>>(qkv, Qb); }
    {   int n = T_SEQ * NKV * 32;
        rope_k_kernel<<<(n + 255) / 256, 256, 0, stream>>>(qkv, Kb); }
    {   int n = T_SEQ * NKV * DH;
        pack_v_kernel<<<(n + 255) / 256, 256, 0, stream>>>(qkv, Vtb); }

    attn_wmma<<<dim3(T_SEQ / 64, NH), 128, 0, stream>>>(Qb, Kb, Vtb, attn);

    // output projection -> fp32 d_out
    gemm_bf16_lds<<<dim3(C_DIM / 128, T_SEQ / 128), 256, 0, stream>>>(
        attn, woT, out, C_DIM, C_DIM);
}
